// FlashSVDRoPEAttention_11252814316332
// MI455X (gfx1250) — matrix operand order repaired and checked by reading, not compiled, not run on hardware
//
#include <hip/hip_runtime.h>

#define BATCH 4
#define SEQ 2048
#define RANK 256
#define NH 16
#define DHD 64
#define DMODEL 1024

typedef __attribute__((ext_vector_type(16))) __bf16 v16bf;
typedef __attribute__((ext_vector_type(8)))  __bf16 v8bf;
typedef __attribute__((ext_vector_type(8)))  float  v8f;

union bf16x16 { v16bf v; v8bf h[2]; };

// ---------------- WMMA fragment helpers (CDNA5 wave32 layouts) ----------------

// A-matrix 16x32 bf16 (M x K), row-major source with row stride `ld`.
// Per lane: two contiguous 8-element groups -> 2x global_load_b128.
__device__ __forceinline__ v16bf load_a_frag(const __bf16* __restrict__ base, int ld, int lane) {
  const int row  = lane & 15;
  const int koff = (lane >> 4) << 4;           // +16 for upper half-wave
  const __bf16* p = base + (size_t)row * ld + koff;
  bf16x16 u;
  u.h[0] = *(const v8bf*)(p);                  // K = koff + 0..7   (lower half)
  u.h[1] = *(const v8bf*)(p + 8);              // K = koff + 8..15  (upper half)
  return u.v;
}

// B-matrix 32x16 where logical B[k][n] = src[n][k] (i.e. K^T), row stride `ld`.
// Per lane: 16 contiguous elements -> 2x global_load_b128.
__device__ __forceinline__ v16bf load_bT_frag(const __bf16* __restrict__ base, int ld, int lane) {
  const int n    = lane & 15;
  const int koff = (lane >> 4) << 4;
  return *(const v16bf*)(base + (size_t)n * ld + koff);
}

// B-matrix fragment from pre-swizzled storage: lane l's 16 elements contiguous.
__device__ __forceinline__ v16bf load_b_frag_sw(const __bf16* __restrict__ frag_base, int lane) {
  return *(const v16bf*)(frag_base + (size_t)lane * 16);
}

__device__ __forceinline__ v8f wmma_bf16(v16bf a, v16bf b, v8f c) {
  return __builtin_amdgcn_wmma_f32_16x16x32_bf16(false, a, false, b, (short)0, c, false, false);
}

// ---------------- Stage 0: precision conversion + swizzle + mask bias ----------------

__global__ void conv_bf16_kernel(const float* __restrict__ src, __bf16* __restrict__ dst, int n) {
  int i = blockIdx.x * blockDim.x + threadIdx.x;
  if (i < n) dst[i] = (__bf16)src[i];
}

// Swizzle SVD factor [R][D] into B-fragment layout:
//   [R/32 chunks][D/16 tiles][32 lanes][16 elems], lane = (k&16) | (d&15), elem = k&15.
__global__ void conv_w_swizzle_kernel(const float* __restrict__ src, __bf16* __restrict__ dst, int n) {
  int i = blockIdx.x * blockDim.x + threadIdx.x;
  if (i >= n) return;
  int k = i / DMODEL, dcol = i % DMODEL;
  int chunk = k >> 5, kin = k & 31;
  int tile  = dcol >> 4;
  int lane  = ((kin >> 4) << 4) | (dcol & 15);
  int idx   = kin & 15;
  size_t off = (((size_t)chunk * (DMODEL / 16) + tile) * 32 + lane) * 16 + idx;
  dst[off] = (__bf16)src[i];
}

__global__ void mask_kernel(const int* __restrict__ m, float* __restrict__ mb, int n) {
  int i = blockIdx.x * blockDim.x + threadIdx.x;
  if (i < n) mb[i] = m[i] ? 0.0f : -__builtin_inff();
}

// ---------------- Stage 1: low-rank projection + bias + RoPE ----------------
// Each wave computes a 16(m) x 64(d) tile of Q, K, or V. 64 cols == one head.
// Q/K stored row-major [B][M][D]; V stored directly in attention B-fragment layout.

__global__ __launch_bounds__(256)
void svd_proj_rope_kernel(const __bf16* __restrict__ Pbf,  // [3][B][M][R]
                          const __bf16* __restrict__ Wsw,  // [3][R/32][D/16][32][16]
                          const float* __restrict__ bq,
                          const float* __restrict__ bk,
                          const float* __restrict__ bv,
                          const int* __restrict__ pos_ids, // [B][M]
                          __bf16* __restrict__ QK,         // [2][B][M][D]
                          __bf16* __restrict__ Vsw)        // [B][NH][SEQ/32][4][32][16]
{
  const int lane = threadIdx.x & 31;
  const int wid  = blockIdx.x * 8 + (threadIdx.x >> 5);
  const int dt   = wid & 15;
  const int mt   = (wid >> 4) & 127;
  const int b    = (wid >> 11) & 3;
  const int sel  = wid >> 13;                  // 0=Q 1=K 2=V
  const int m0   = mt * 16;
  const int d0   = dt * 64;
  const int ncol = lane & 15;
  const int hsel = lane >> 4;

  const __bf16* P = Pbf + ((size_t)sel * BATCH * SEQ + (size_t)b * SEQ + m0) * RANK;
  const __bf16* W = Wsw + (size_t)sel * (RANK / 32) * (DMODEL / 16) * 32 * 16;

  v8f acc[4] = {};
#pragma unroll
  for (int k = 0; k < RANK; k += 32) {
    v16bf a = load_a_frag(P + k, RANK, lane);
    const __bf16* Wc = W + (((size_t)(k >> 5) * (DMODEL / 16) + (d0 >> 4)) * 32) * 16;
#pragma unroll
    for (int t = 0; t < 4; ++t) {
      v16bf bfr = load_b_frag_sw(Wc + (size_t)t * 32 * 16, lane);
      acc[t] = wmma_bf16(a, bfr, acc[t]);
    }
  }

  const float* bias = (sel == 0) ? bq : (sel == 1) ? bk : bv;
#pragma unroll
  for (int t = 0; t < 4; ++t) {
    float bb = bias[d0 + t * 16 + ncol];
#pragma unroll
    for (int g = 0; g < 8; ++g) acc[t][g] += bb;
  }

  if (sel < 2) {
    // RoPE half-rotation: pair (i, i+32) inside the 64-wide head.
#pragma unroll
    for (int t = 0; t < 2; ++t) {
      int i = t * 16 + ncol;                            // 0..31 frequency index
      float invf = __expf(-(2.0f * (float)i / 64.0f) * 9.210340371976184f); // 10000^(-2i/64)
#pragma unroll
      for (int g = 0; g < 8; ++g) {
        int row   = m0 + g + (hsel << 3);
        float pos = (float)pos_ids[b * SEQ + row];
        float f = pos * invf, sn, cs;
        __sincosf(f, &sn, &cs);
        float x0 = acc[t][g], x1 = acc[t + 2][g];
        acc[t][g]     = x0 * cs - x1 * sn;
        acc[t + 2][g] = x0 * sn + x1 * cs;
      }
    }
    __bf16* O = QK + ((size_t)sel * BATCH * SEQ + (size_t)b * SEQ + m0) * DMODEL + d0;
#pragma unroll
    for (int t = 0; t < 4; ++t)
#pragma unroll
      for (int g = 0; g < 8; ++g) {
        int row = g + (hsel << 3);
        O[(size_t)row * DMODEL + t * 16 + ncol] = (__bf16)acc[t][g];
      }
  } else {
    // Scatter V directly into attention B-fragment layout.
    // element (key, dh): lane' = (key&16) | (dh&15), elem = key&15
#pragma unroll
    for (int t = 0; t < 4; ++t)
#pragma unroll
      for (int g = 0; g < 8; ++g) {
        int key   = m0 + g + (hsel << 3);
        int chunk = key >> 5;
        int kin   = key & 31;
        int lp    = ((kin >> 4) << 4) | ncol;
        size_t off = (((((size_t)b * NH + dt) * (SEQ / 32) + chunk) * 4 + t) * 32 + lp) * 16 + (kin & 15);
        Vsw[off] = (__bf16)acc[t][g];
      }
  }
}

// ---------------- Stage 2: flash attention with online softmax ----------------
// 8 waves/block; each wave owns 32 query rows (two 16-row tiles) of one (b, h),
// so every K^T / V fragment load feeds 2x the WMMAs. Keys in chunks of 32.

__global__ __launch_bounds__(256)
void flash_attn_kernel(const __bf16* __restrict__ Q,      // [B][M][D] (RoPE applied)
                       const __bf16* __restrict__ K,      // [B][M][D] (RoPE applied)
                       const __bf16* __restrict__ Vsw,    // [B][NH][SEQ/32][4][32][16]
                       const float* __restrict__ maskbias,// [B][M] (0 or -inf)
                       float* __restrict__ out)           // [B][M][H][DHD] fp32
{
  __shared__ float pbuf[8][32][33];   // per-wave D-layout -> A-layout transpose (2 row tiles)
  const int lane = threadIdx.x & 31;
  const int w    = threadIdx.x >> 5;
  const int tile = blockIdx.x;
  const int mblk = tile & 7;          // SEQ/256 = 8
  const int h    = (tile >> 3) & 15;
  const int b    = tile >> 7;
  const int m0   = mblk * 256 + w * 32;
  const int ncol = lane & 15;
  const int hsel = lane >> 4;
  const float scale = 0.125f;         // 1/sqrt(64)

  // Two 16-row query tiles per wave, each needs dh 0..31 and 32..63 A-fragments.
  v16bf qa[2][2];
#pragma unroll
  for (int rt = 0; rt < 2; ++rt) {
    const __bf16* Qb = Q + ((size_t)b * SEQ + m0 + rt * 16) * DMODEL + h * DHD;
    qa[rt][0] = load_a_frag(Qb, DMODEL, lane);
    qa[rt][1] = load_a_frag(Qb + 32, DMODEL, lane);
  }

  const __bf16* Vh  = Vsw + (((size_t)b * NH + h) * (SEQ / 32)) * 4 * 32 * 16;
  const float* mrow = maskbias + b * SEQ;

  v8f o[2][4] = {};
  float mst[2][8], lst[2][8];
#pragma unroll
  for (int rt = 0; rt < 2; ++rt)
#pragma unroll
    for (int g = 0; g < 8; ++g) { mst[rt][g] = -__builtin_inff(); lst[rt][g] = 0.f; }

  for (int n0 = 0; n0 < SEQ; n0 += 32) {
    const __bf16* Kb = K + ((size_t)b * SEQ + n0) * DMODEL + h * DHD;
    const __bf16* Vc = Vh + (size_t)(n0 >> 5) * 4 * 32 * 16;
    if (n0 + 32 < SEQ) {               // global_prefetch_b8 for next chunk
      __builtin_prefetch(Kb + (size_t)32 * DMODEL, 0, 1);
      __builtin_prefetch(Vc + (size_t)4 * 32 * 16, 0, 1);
    }

    // K^T fragments for both 16-key subtiles, dh halves (shared by both row tiles)
    v16bf kb[2][2];
#pragma unroll
    for (int sN = 0; sN < 2; ++sN) {
      kb[sN][0] = load_bT_frag(Kb + (size_t)sN * 16 * DMODEL, DMODEL, lane);
      kb[sN][1] = load_bT_frag(Kb + (size_t)sN * 16 * DMODEL + 32, DMODEL, lane);
    }
    float mb0 = mrow[n0 + ncol];
    float mb1 = mrow[n0 + 16 + ncol];

#pragma unroll
    for (int rt = 0; rt < 2; ++rt) {
      // scores for this row tile: 2 N-subtiles, dh=64 via 2 chained WMMAs each
      v8f s[2];
#pragma unroll
      for (int sN = 0; sN < 2; ++sN) {
        v8f z = {};
        z = wmma_bf16(qa[rt][0], kb[sN][0], z);
        z = wmma_bf16(qa[rt][1], kb[sN][1], z);
        float mb = (sN == 0) ? mb0 : mb1;
#pragma unroll
        for (int g = 0; g < 8; ++g) s[sN][g] = z[g] * scale + mb;
      }

      // online softmax per row (row = g + 8*hsel; reduce across 16-lane N groups)
#pragma unroll
      for (int g = 0; g < 8; ++g) {
        float v = fmaxf(s[0][g], s[1][g]);
#pragma unroll
        for (int off = 8; off >= 1; off >>= 1) v = fmaxf(v, __shfl_xor(v, off, 16));
        float mnew  = fmaxf(mst[rt][g], v);
        float corr  = (mst[rt][g] == mnew) ? 1.0f : __expf(mst[rt][g] - mnew);
        float msafe = (mnew == -__builtin_inff()) ? 0.0f : mnew;
        float a0 = __expf(s[0][g] - msafe);
        float a1 = __expf(s[1][g] - msafe);
        float rs = a0 + a1;
#pragma unroll
        for (int off = 8; off >= 1; off >>= 1) rs += __shfl_xor(rs, off, 16);
        lst[rt][g] = lst[rt][g] * corr + rs;
        mst[rt][g] = mnew;
#pragma unroll
        for (int t = 0; t < 4; ++t) o[rt][t][g] *= corr;
        // stage P into LDS transpose buffer (D-layout -> A-layout)
        int row = rt * 16 + g + (hsel << 3);
        pbuf[w][row][ncol]      = a0;
        pbuf[w][row][16 + ncol] = a1;
      }
    }

    asm volatile("s_wait_dscnt 0" ::: "memory");

    // read both P tiles as A-fragments
    v16bf pa[2];
#pragma unroll
    for (int rt = 0; rt < 2; ++rt) {
      const int row  = rt * 16 + ncol;
      const int koff = hsel << 4;
#pragma unroll
      for (int g = 0; g < 8; ++g) {
        int k = (g << 1) + koff;
        pa[rt][2 * g]     = (__bf16)pbuf[w][row][k];
        pa[rt][2 * g + 1] = (__bf16)pbuf[w][row][k + 1];
      }
    }

    // O += P[16x32] * V[32x64], V fragments shared by both row tiles
#pragma unroll
    for (int t = 0; t < 4; ++t) {
      v16bf vb = load_b_frag_sw(Vc + (size_t)t * 32 * 16, lane);
      o[0][t] = wmma_bf16(pa[0], vb, o[0][t]);
      o[1][t] = wmma_bf16(pa[1], vb, o[1][t]);
    }
  }

  // epilogue: divide by running sum; fully-masked rows -> 0
#pragma unroll
  for (int rt = 0; rt < 2; ++rt)
#pragma unroll
    for (int g = 0; g < 8; ++g) {
      int row = m0 + rt * 16 + g + (hsel << 3);
      float l = lst[rt][g];
      float inv = (l > 0.f) ? 1.0f / l : 0.0f;
      float* op = out + (((size_t)b * SEQ + row) * NH + h) * DHD;
#pragma unroll
      for (int t = 0; t < 4; ++t) op[t * 16 + ncol] = o[rt][t][g] * inv;
    }
}

// ---------------- Host launch ----------------

extern "C" void kernel_launch(void* const* d_in, const int* in_sizes, int n_in,
                              void* d_out, int out_size, void* d_ws, size_t ws_size,
                              hipStream_t stream) {
  const float* Pq = (const float*)d_in[0];
  const float* Pk = (const float*)d_in[1];
  const float* Pv = (const float*)d_in[2];
  const float* Vq = (const float*)d_in[3];
  const float* Vk = (const float*)d_in[4];
  const float* Vv = (const float*)d_in[5];
  const float* bq = (const float*)d_in[6];
  const float* bk = (const float*)d_in[7];
  const float* bv = (const float*)d_in[8];
  const int* amask = (const int*)d_in[9];
  const int* pos   = (const int*)d_in[10];
  float* out = (float*)d_out;

  char* w = (char*)d_ws;
  size_t off = 0;
  __bf16* Pbf = (__bf16*)(w + off); off += (size_t)3 * BATCH * SEQ * RANK * sizeof(__bf16);
  __bf16* Wsw = (__bf16*)(w + off); off += (size_t)3 * RANK * DMODEL * sizeof(__bf16);
  __bf16* QK  = (__bf16*)(w + off); off += (size_t)2 * BATCH * SEQ * DMODEL * sizeof(__bf16);
  __bf16* Vsw = (__bf16*)(w + off); off += (size_t)BATCH * SEQ * DMODEL * sizeof(__bf16);
  float*  mb  = (float*)(w + off);  off += (size_t)BATCH * SEQ * sizeof(float);
  (void)ws_size; (void)in_sizes; (void)n_in; (void)out_size;

  const int PN = BATCH * SEQ * RANK;
  const int VN = RANK * DMODEL;
  conv_bf16_kernel<<<(PN + 255) / 256, 256, 0, stream>>>(Pq, Pbf, PN);
  conv_bf16_kernel<<<(PN + 255) / 256, 256, 0, stream>>>(Pk, Pbf + (size_t)PN, PN);
  conv_bf16_kernel<<<(PN + 255) / 256, 256, 0, stream>>>(Pv, Pbf + (size_t)2 * PN, PN);
  conv_w_swizzle_kernel<<<(VN + 255) / 256, 256, 0, stream>>>(Vq, Wsw, VN);
  conv_w_swizzle_kernel<<<(VN + 255) / 256, 256, 0, stream>>>(Vk, Wsw + (size_t)VN, VN);
  conv_w_swizzle_kernel<<<(VN + 255) / 256, 256, 0, stream>>>(Vv, Wsw + (size_t)2 * VN, VN);
  mask_kernel<<<(BATCH * SEQ + 255) / 256, 256, 0, stream>>>(amask, mb, BATCH * SEQ);

  // 3 mats * 4 batches * 128 m-tiles * 16 d-tiles = 24576 wave tiles / 8 waves = 3072 blocks
  svd_proj_rope_kernel<<<3072, 256, 0, stream>>>(Pbf, Wsw, bq, bk, bv, pos, QK, Vsw);

  const __bf16* Qs = QK;
  const __bf16* Ks = QK + (size_t)BATCH * SEQ * DMODEL;
  // 4 batches * 16 heads * 8 m-blocks (256 rows each, 32 per wave) = 512 blocks
  flash_attn_kernel<<<BATCH * NH * (SEQ / 256), 256, 0, stream>>>(Qs, Ks, Vsw, mb, out);
}
